// CausalSelfAttention_26706106646552
// MI455X (gfx1250) — matrix operand order repaired and checked by reading, not compiled, not run on hardware
//
#include <hip/hip_runtime.h>
#include <hip/hip_bf16.h>
#include <math.h>

#define Bz  2
#define Tz  2048
#define Cz  2048
#define Hz  16
#define HDz 128

typedef __attribute__((ext_vector_type(16))) __bf16 v16bf;
typedef __attribute__((ext_vector_type(8)))  float  v8f;
typedef __attribute__((ext_vector_type(4)))  int    v4i;

union BFrag { v16bf v; unsigned short u[16]; };

__device__ __forceinline__ unsigned short f2bf(float f) {
    unsigned int u = __float_as_uint(f);
    unsigned int r = u + 0x7FFFu + ((u >> 16) & 1u);   // round-to-nearest-even
    return (unsigned short)(r >> 16);
}

// ---------------------------------------------------------------------------
// Async global->LDS 16-byte copy (gfx1250 GLOBAL_LOAD_ASYNC_TO_LDS_B128),
// with a synchronous fallback if the builtin is unavailable.
// ---------------------------------------------------------------------------
#if defined(__gfx1250__) && __has_builtin(__builtin_amdgcn_global_load_async_to_lds_b128)
#define ASYNC_LDS 1
#endif

__device__ __forceinline__ void async_copy16(const void* g, void* l) {
#if ASYNC_LDS
    __builtin_amdgcn_global_load_async_to_lds_b128(
        (__attribute__((address_space(1))) v4i*)g,
        (__attribute__((address_space(3))) v4i*)l,
        0, 0);
#else
    *(v4i*)l = *(const v4i*)g;
#endif
}

__device__ __forceinline__ void async_wait() {
#if ASYNC_LDS
#if __has_builtin(__builtin_amdgcn_s_wait_asynccnt)
    __builtin_amdgcn_s_wait_asynccnt(0);
#else
    asm volatile("s_wait_asynccnt 0x0" ::: "memory");
#endif
#endif
}

// ---------------------------------------------------------------------------
// f32 -> bf16 conversion (grid-stride)
// ---------------------------------------------------------------------------
__global__ void f32_to_bf16_kernel(const float* __restrict__ src,
                                   unsigned short* __restrict__ dst, int n) {
    int i = blockIdx.x * blockDim.x + threadIdx.x;
    int stride = gridDim.x * blockDim.x;
    for (; i < n; i += stride) dst[i] = f2bf(src[i]);
}

// ---------------------------------------------------------------------------
// Y[m,n] = sum_k A[m,k] * Wb[n,k] + bias[n]
// A:  bf16 (M x K) row-major.   Wb: bf16 (N x K) row-major.
// Output: f32 (Yf) or bf16 (Yh).  Tile 64x64x32, 4 waves, wave = 16x64 strip.
// Tiles staged via async global->LDS b128 copies.
// ---------------------------------------------------------------------------
__global__ __launch_bounds__(128)
void gemm_bf16_wmma(const unsigned short* __restrict__ A,
                    const unsigned short* __restrict__ Wb,
                    const float* __restrict__ bias,
                    float* __restrict__ Yf,
                    unsigned short* __restrict__ Yh,
                    int M, int N, int K) {
    __shared__ unsigned short As[64][40];   // 80 B row stride: 16B-aligned + pad
    __shared__ unsigned short Bs[64][40];

    const int tid  = threadIdx.x;
    const int lane = tid & 31;
    const int wv   = tid >> 5;
    const int l15  = lane & 15;
    const int hi   = lane >> 4;
    const int n0   = blockIdx.x * 64;
    const int m0   = blockIdx.y * 64;

    v8f acc[4];
#pragma unroll
    for (int s = 0; s < 4; ++s) acc[s] = {};

    for (int k0 = 0; k0 < K; k0 += 32) {
        // Stage 64x32 bf16 tiles of A and W: 256 chunks of 16B each, 2/thread.
#pragma unroll
        for (int i = 0; i < 2; ++i) {
            const int q   = tid * 2 + i;     // 0..255
            const int row = q >> 2;
            const int co  = (q & 3) * 8;     // ushort offset within row
            async_copy16(&A [(size_t)(m0 + row) * K + k0 + co], &As[row][co]);
            async_copy16(&Wb[(size_t)(n0 + row) * K + k0 + co], &Bs[row][co]);
        }
        __builtin_prefetch(&Wb[(size_t)(n0 + (tid >> 1)) * K + k0 + 32], 0, 0);
        async_wait();
        __syncthreads();

        BFrag af;   // A fragment: lane=row, 16 contiguous K per half-wave
        {
            const int r  = wv * 16 + l15;
            const int kb = hi * 16;
#pragma unroll
            for (int j = 0; j < 8; ++j) {
                af.u[j]     = As[r][kb + j];
                af.u[j + 8] = As[r][kb + 8 + j];
            }
        }
#pragma unroll
        for (int s = 0; s < 4; ++s) {
            BFrag bf;   // B fragment: lane=col, 16 contiguous K per half-wave
            const int col = s * 16 + l15;
            const int kb  = hi * 16;
#pragma unroll
            for (int j = 0; j < 16; ++j) bf.u[j] = Bs[col][kb + j];
            acc[s] = __builtin_amdgcn_wmma_f32_16x16x32_bf16(
                false, af.v, false, bf.v, (short)0, acc[s], false, false);
        }
        __syncthreads();
    }

#pragma unroll
    for (int s = 0; s < 4; ++s) {
        const int   colg = n0 + s * 16 + l15;
        const float bv   = bias[colg];
#pragma unroll
        for (int r = 0; r < 8; ++r) {
            const int    mg  = m0 + wv * 16 + hi * 8 + r;
            const float  val = acc[s][r] + bv;
            const size_t idx = (size_t)mg * N + colg;
            if (Yf) Yf[idx] = val;
            else    Yh[idx] = f2bf(val);
        }
    }
}

// ---------------------------------------------------------------------------
// Flash attention (causal, online softmax).  Grid: (T/64, B*H), 128 threads.
// Each wave owns 16 query rows.  V tile is async-staged to LDS and the wait
// is deferred past the QK^T WMMAs and softmax (copy/compute overlap).
// Output written in the reference's (B,H,hd,T)-contiguous layout, bf16.
// ---------------------------------------------------------------------------
__global__ __launch_bounds__(128)
void flash_attn_wmma(const unsigned short* __restrict__ Q,
                     const unsigned short* __restrict__ Km,
                     const unsigned short* __restrict__ V,
                     unsigned short* __restrict__ O) {
    __shared__ unsigned short plds[64][72];
    __shared__ unsigned short vlds[64][136];   // 272 B row stride (16B-aligned)

    const int tid  = threadIdx.x;
    const int lane = tid & 31;
    const int wv   = tid >> 5;
    const int l15  = lane & 15;
    const int hi   = lane >> 4;
    const int qt   = blockIdx.x;
    const int bh   = blockIdx.y;
    const int b    = bh / Hz;
    const int h    = bh % Hz;
    const int q0   = qt * 64 + wv * 16;                 // wave's first query
    const size_t base = (size_t)b * Tz * Cz + (size_t)h * HDz;

    v8f o[8];
#pragma unroll
    for (int s = 0; s < 8; ++s) o[s] = {};
    float mrow[8], lrow[8];
#pragma unroll
    for (int r = 0; r < 8; ++r) { mrow[r] = -1e30f; lrow[r] = 0.0f; }

    // Preload Q A-fragments for all 4 K-chunks (head_dim 128 = 4 x 32)
    BFrag aq[4];
    {
        const unsigned short* qp = &Q[base + (size_t)(q0 + l15) * Cz];
        const int kb = hi * 16;
#pragma unroll
        for (int c = 0; c < 4; ++c)
#pragma unroll
            for (int j = 0; j < 8; ++j) {
                aq[c].u[j]     = qp[c * 32 + kb + j];
                aq[c].u[j + 8] = qp[c * 32 + kb + 8 + j];
            }
    }
    const float scale = 0.08838834764831845f;   // 1/sqrt(128)

    for (int kt = 0; kt <= qt; ++kt) {
        const int k0 = kt * 64;

        // ---- kick off async V-tile copy: 64 keys x 128 dims bf16 = 16 KB ----
#pragma unroll
        for (int i = 0; i < 8; ++i) {
            const int q   = tid * 8 + i;     // 0..1023
            const int row = q >> 4;          // key within tile
            const int co  = (q & 15) * 8;    // ushort offset within row
            async_copy16(&V[base + (size_t)(k0 + row) * Cz + co], &vlds[row][co]);
        }

        // ---- S = Q @ K^T  (16 x 64), 4 N-subtiles x 4 K-chunks ----
        v8f sacc[4];
#pragma unroll
        for (int s = 0; s < 4; ++s) sacc[s] = {};
#pragma unroll
        for (int s = 0; s < 4; ++s) {
            const int key = k0 + s * 16 + l15;
            const unsigned short* kp = &Km[base + (size_t)key * Cz + hi * 16];
#pragma unroll
            for (int c = 0; c < 4; ++c) {
                BFrag bf;
#pragma unroll
                for (int j = 0; j < 16; ++j) bf.u[j] = kp[c * 32 + j];
                sacc[s] = __builtin_amdgcn_wmma_f32_16x16x32_bf16(
                    false, aq[c].v, false, bf.v, (short)0, sacc[s], false, false);
            }
        }

        // ---- online softmax (rows live in 16-lane halves) ----
#pragma unroll
        for (int r = 0; r < 8; ++r) {
            const int qrow = q0 + hi * 8 + r;
            float mx = -1e30f;
#pragma unroll
            for (int s = 0; s < 4; ++s) {
                const int   col = k0 + s * 16 + l15;
                const float sv  = (col <= qrow) ? sacc[s][r] * scale : -1e30f;
                sacc[s][r] = sv;
                mx = fmaxf(mx, sv);
            }
#pragma unroll
            for (int msk = 8; msk >= 1; msk >>= 1)
                mx = fmaxf(mx, __shfl_xor(mx, msk, 32));
            const float mnew  = fmaxf(mrow[r], mx);
            const float alpha = __expf(mrow[r] - mnew);
            float rsum = 0.0f;
#pragma unroll
            for (int s = 0; s < 4; ++s) {
                const float p = __expf(sacc[s][r] - mnew);
                sacc[s][r] = p;
                rsum += p;
            }
#pragma unroll
            for (int msk = 8; msk >= 1; msk >>= 1)
                rsum += __shfl_xor(rsum, msk, 32);
            lrow[r] = lrow[r] * alpha + rsum;
            mrow[r] = mnew;
#pragma unroll
            for (int s2 = 0; s2 < 8; ++s2) o[s2][r] *= alpha;
        }

        // ---- P: D-layout -> A-layout via LDS ----
#pragma unroll
        for (int s = 0; s < 4; ++s)
#pragma unroll
            for (int r = 0; r < 8; ++r)
                plds[wv * 16 + hi * 8 + r][s * 16 + l15] = f2bf(sacc[s][r]);

        async_wait();       // V tile resident in LDS
        __syncthreads();    // P + V visible to all waves

        BFrag ap[2];   // P as A-matrix: 16 x 64 keys = 2 chunks of K=32
        {
            const int prow = wv * 16 + l15;
            const int kb   = hi * 16;
#pragma unroll
            for (int c = 0; c < 2; ++c)
#pragma unroll
                for (int j = 0; j < 8; ++j) {
                    ap[c].u[j]     = plds[prow][c * 32 + kb + j];
                    ap[c].u[j + 8] = plds[prow][c * 32 + kb + 8 + j];
                }
        }

        // ---- O += P @ V  (8 d-subtiles; B frag from LDS, strided by key) ----
#pragma unroll
        for (int s = 0; s < 8; ++s) {
            const int dcol = s * 16 + l15;
#pragma unroll
            for (int c = 0; c < 2; ++c) {
                BFrag bfv;
                const int keyb = c * 32 + hi * 16;
#pragma unroll
                for (int j = 0; j < 16; ++j)
                    bfv.u[j] = vlds[keyb + j][dcol];
                o[s] = __builtin_amdgcn_wmma_f32_16x16x32_bf16(
                    false, ap[c].v, false, bfv.v, (short)0, o[s], false, false);
            }
        }
        __syncthreads();    // done with plds/vlds before next tile overwrites
    }

    // ---- normalize, store in (B,H,hd,T)-contiguous layout ----
#pragma unroll
    for (int s = 0; s < 8; ++s) {
        const int d = s * 16 + l15;
#pragma unroll
        for (int r = 0; r < 8; ++r) {
            const int   qrow = q0 + hi * 8 + r;
            const float val  = o[s][r] / lrow[r];
            O[((size_t)bh * HDz + d) * Tz + qrow] = f2bf(val);
        }
    }
}

// ---------------------------------------------------------------------------
extern "C" void kernel_launch(void* const* d_in, const int* in_sizes, int n_in,
                              void* d_out, int out_size, void* d_ws, size_t ws_size,
                              hipStream_t stream) {
    const float* x  = (const float*)d_in[0];
    const float* Wq = (const float*)d_in[1];
    const float* bq = (const float*)d_in[2];
    const float* Wk = (const float*)d_in[3];
    const float* bk = (const float*)d_in[4];
    const float* Wv = (const float*)d_in[5];
    const float* bv = (const float*)d_in[6];
    const float* Wp = (const float*)d_in[7];
    const float* bp = (const float*)d_in[8];

    const int    M  = Bz * Tz;                 // 4096
    const size_t NM = (size_t)M * Cz;          // 8,388,608 elems
    const size_t NW = (size_t)Cz * Cz;         // 4,194,304 elems

    unsigned short* xb  = (unsigned short*)d_ws;     // x in bf16
    unsigned short* qb  = xb  + NM;
    unsigned short* kb  = qb  + NM;
    unsigned short* vb  = kb  + NM;
    unsigned short* ab  = vb  + NM;                  // attn out, buggy layout
    unsigned short* wqb = ab  + NM;                  // weights in bf16
    unsigned short* wkb = wqb + NW;
    unsigned short* wvb = wkb + NW;
    unsigned short* wpb = wvb + NW;

    f32_to_bf16_kernel<<<2048, 256, 0, stream>>>(x,  xb,  (int)NM);
    f32_to_bf16_kernel<<<1024, 256, 0, stream>>>(Wq, wqb, (int)NW);
    f32_to_bf16_kernel<<<1024, 256, 0, stream>>>(Wk, wkb, (int)NW);
    f32_to_bf16_kernel<<<1024, 256, 0, stream>>>(Wv, wvb, (int)NW);
    f32_to_bf16_kernel<<<1024, 256, 0, stream>>>(Wp, wpb, (int)NW);

    dim3 g(Cz / 64, M / 64);                   // (32, 64)
    gemm_bf16_wmma<<<g, 128, 0, stream>>>(xb, wqb, bq, nullptr, qb, M, Cz, Cz);
    gemm_bf16_wmma<<<g, 128, 0, stream>>>(xb, wkb, bk, nullptr, kb, M, Cz, Cz);
    gemm_bf16_wmma<<<g, 128, 0, stream>>>(xb, wvb, bv, nullptr, vb, M, Cz, Cz);

    flash_attn_wmma<<<dim3(Tz / 64, Bz * Hz), 128, 0, stream>>>(qb, kb, vb, ab);

    gemm_bf16_wmma<<<g, 128, 0, stream>>>(ab, wpb, bp, (float*)d_out, nullptr,
                                          M, Cz, Cz);
}
